// VectorQuantizer_73735998538496
// MI455X (gfx1250) — matrix operand order repaired and checked by reading, not compiled, not run on hardware
//
#include <hip/hip_runtime.h>
#include <hip/hip_bf16.h>

// CDNA5 / gfx1250 VQ kernel.
// GEMM term -2*f@C^T done with v_wmma_f32_16x16x32_bf16 using a bf16 hi/lo
// split of both operands (3 WMMAs per k-step) for ~fp32 accuracy at matrix
// throughput. Packed codebook (WMMA B layout, hi+lo) lives in 320KB WGP LDS.
// Roofline: ~34 MB HBM traffic (z in + z_q out) => ~1.5us floor @ 23.3 TB/s;
// 1.57M wmma instrs total keeps the matrix pipes well under that floor.

typedef __attribute__((ext_vector_type(16))) __bf16 v16bf;
typedef __attribute__((ext_vector_type(8)))  float  v8f;

struct U8 { unsigned int d[8]; };

__device__ __forceinline__ unsigned int f2bf_rne(float f) {
    unsigned int x = __float_as_uint(f);
    return (x + 0x7FFFu + ((x >> 16) & 1u)) >> 16;   // round-to-nearest-even
}
__device__ __forceinline__ float bf_as_f32(unsigned int h) {
    return __uint_as_float(h << 16);
}
// order-preserving fp32 -> u32 map (NaN-free inputs)
__device__ __forceinline__ unsigned int fkey(float f) {
    unsigned int b = __float_as_uint(f);
    return b ^ ((unsigned int)((int)b >> 31) | 0x80000000u);
}

#define WAVES          8
#define THREADS        256
#define NCODE          1024
#define DIMS           64
#define CODE_TILES     64          // 1024 / 16
#define NUM_WGS        256
#define TILES_PER_WG   16          // 4096 token tiles / 256 WGs
#define TILES_PER_WAVE 2
#define ZROW           68          // padded LDS row stride (floats), bank-spread
#define NTOT           4194304     // 65536 * 64

__global__ __launch_bounds__(THREADS, 1)
void vq_main(const float* __restrict__ z, const float* __restrict__ cb,
             float* __restrict__ out, float* __restrict__ wgPartial)
{
    // ---- LDS: 256KB packed codebook + 4KB norms + 34KB z staging ----
    __shared__ unsigned int Bpack[65536];          // [t][h2][hl][p][lane] dwords
    __shared__ float cnormsh[NCODE];
    __shared__ float zsh[WAVES * 16 * ZROW];
    __shared__ int   idxbuf[WAVES * 16];
    __shared__ float lossW[WAVES];

    const int tid  = threadIdx.x;
    const int wid  = tid >> 5;
    const int lane = tid & 31;
    const int half = lane >> 4;
    const int ln   = lane & 15;

    // ================= pack codebook into LDS (bf16 hi/lo, B layout) ======
    // B layout (16-bit, 32x16): VGPR p holds K pair at
    // (p/4)*16 + half*8 + (p%4)*2 (+0,+1); col n = lane%16.
    #pragma unroll 4
    for (int j = 0; j < 65536 / THREADS; ++j) {
        int o     = tid + THREADS * j;
        int olane = o & 31;
        int p     = (o >> 5) & 7;
        int hl    = (o >> 8) & 1;    // 0 = hi, 1 = lo
        int h2    = (o >> 9) & 1;    // which K-half of D=64
        int t     = o >> 10;         // code tile 0..63
        int code  = t * 16 + (olane & 15);
        int d0    = 32 * h2 + ((p >> 2) << 4) + (olane >> 4) * 8 + (p & 3) * 2;
        float f0 = cb[code * DIMS + d0];
        float f1 = cb[code * DIMS + d0 + 1];
        unsigned int h0 = f2bf_rne(f0), h1 = f2bf_rne(f1);
        unsigned int v0, v1;
        if (hl == 0) { v0 = h0; v1 = h1; }
        else { v0 = f2bf_rne(f0 - bf_as_f32(h0)); v1 = f2bf_rne(f1 - bf_as_f32(h1)); }
        Bpack[o] = (v1 << 16) | v0;
    }
    // codebook squared norms
    for (int c = tid; c < NCODE; c += THREADS) {
        const float4* r4 = (const float4*)(cb + c * DIMS);
        float s = 0.f;
        #pragma unroll
        for (int k = 0; k < 16; ++k) {
            float4 q = r4[k];
            s += q.x * q.x + q.y * q.y + q.z * q.z + q.w * q.w;
        }
        cnormsh[c] = s;
    }
    __syncthreads();

    float lossAcc = 0.f;

    for (int it = 0; it < TILES_PER_WAVE; ++it) {
        const int tile = blockIdx.x * TILES_PER_WG + it * WAVES + wid;
        const int row0 = tile * 16;
        float* zs = zsh + wid * 16 * ZROW;

        // ---- stage 16x64 z tile into LDS (padded rows) ----
        {
            const float4* zg4 = (const float4*)(z + (size_t)row0 * DIMS);
            float4* zs4 = (float4*)zs;
            #pragma unroll
            for (int q = 0; q < 8; ++q) {
                int e = lane + 32 * q;            // 0..255 float4s
                zs4[(e >> 4) * (ZROW / 4) + (e & 15)] = zg4[e];
            }
        }
        __syncthreads();

        // ---- build A fragments (bf16 hi/lo), WMMA A layout ----
        // A layout (16-bit, 16x32): row m = lane%16;
        // VGPR p holds K pair at (p/4)*16 + (lane/16)*8 + (p%4)*2 (+0,+1).
        v16bf aHi[2], aLo[2];
        #pragma unroll
        for (int h2 = 0; h2 < 2; ++h2) {
            U8 uh, ul;
            #pragma unroll
            for (int p = 0; p < 8; ++p) {
                int kb = 32 * h2 + ((p >> 2) << 4) + (half << 3) + ((p & 3) << 1);
                const float2* zr = (const float2*)(zs + ln * ZROW);
                float2 f = zr[kb >> 1];
                unsigned int h0 = f2bf_rne(f.x), h1 = f2bf_rne(f.y);
                unsigned int l0 = f2bf_rne(f.x - bf_as_f32(h0));
                unsigned int l1 = f2bf_rne(f.y - bf_as_f32(h1));
                uh.d[p] = (h1 << 16) | h0;
                ul.d[p] = (l1 << 16) | l0;
            }
            aHi[h2] = __builtin_bit_cast(v16bf, uh);
            aLo[h2] = __builtin_bit_cast(v16bf, ul);
        }

        // ---- sweep all 64 code tiles, track running argmin ----
        float bestS[8]; int bestI[8];
        #pragma unroll
        for (int v = 0; v < 8; ++v) { bestS[v] = 3.4e38f; bestI[v] = 0; }

        #pragma unroll 2
        for (int t = 0; t < CODE_TILES; ++t) {
            v8f acc = {};
            #pragma unroll
            for (int h2 = 0; h2 < 2; ++h2) {
                U8 ubh, ubl;
                const unsigned int* bp = &Bpack[(((t * 2 + h2) * 2 + 0) * 8) * 32 + lane];
                #pragma unroll
                for (int p = 0; p < 8; ++p) ubh.d[p] = bp[p * 32];
                const unsigned int* bq = &Bpack[(((t * 2 + h2) * 2 + 1) * 8) * 32 + lane];
                #pragma unroll
                for (int p = 0; p < 8; ++p) ubl.d[p] = bq[p * 32];
                v16bf bh = __builtin_bit_cast(v16bf, ubh);
                v16bf bl = __builtin_bit_cast(v16bf, ubl);
                acc = __builtin_amdgcn_wmma_f32_16x16x32_bf16(false, aHi[h2], false, bh,
                                                              (short)0, acc, false, false);
                acc = __builtin_amdgcn_wmma_f32_16x16x32_bf16(false, aLo[h2], false, bh,
                                                              (short)0, acc, false, false);
                acc = __builtin_amdgcn_wmma_f32_16x16x32_bf16(false, aHi[h2], false, bl,
                                                              (short)0, acc, false, false);
            }
            // score = ||c||^2 - 2 f.c  (||f||^2 dropped: constant per row)
            float cn  = cnormsh[t * 16 + ln];
            int  code = t * 16 + ln;
            #pragma unroll
            for (int v = 0; v < 8; ++v) {
                float s = cn - 2.0f * acc[v];        // row m = v + 8*half, col = ln
                if (s < bestS[v]) { bestS[v] = s; bestI[v] = code; }
            }
        }

        // ---- branchless argmin across the 16 lanes of each half ----
        // pack (order-preserving score key << 32 | code index); u64 min gives
        // min score with first-occurrence (smallest index) tie-break.
        if (ln == 0) {
            // placeholder to keep barrier structure uniform (filled below)
        }
        #pragma unroll
        for (int v = 0; v < 8; ++v) {
            unsigned long long key =
                ((unsigned long long)fkey(bestS[v]) << 32) | (unsigned int)bestI[v];
            #pragma unroll
            for (int m = 1; m < 16; m <<= 1) {
                unsigned int khi = __shfl_xor((unsigned int)(key >> 32), m, 32);
                unsigned int klo = __shfl_xor((unsigned int)(key & 0xFFFFFFFFu), m, 32);
                unsigned long long ok = ((unsigned long long)khi << 32) | klo;
                key = (ok < key) ? ok : key;
            }
            bestI[v] = (int)(unsigned int)(key & 0xFFFFFFFFu);
        }
        if (ln == 0) {
            #pragma unroll
            for (int v = 0; v < 8; ++v) idxbuf[wid * 16 + half * 8 + v] = bestI[v];
        }
        __syncthreads();

        // ---- gather winning codebook rows, write z_q_out, accumulate loss ----
        {
            int r = lane >> 1, hd = lane & 1;        // 2 lanes per token row
            int kidx = idxbuf[wid * 16 + r];
            const float4* c4 = (const float4*)(cb + (size_t)kidx * DIMS + hd * 32);
            const float4* z4 = (const float4*)(zs + r * ZROW + hd * 32);
            float4* o4 = (float4*)(out + (size_t)(row0 + r) * DIMS + hd * 32);
            #pragma unroll
            for (int k = 0; k < 8; ++k) {
                float4 c = c4[k], zv = z4[k];
                o4[k] = c;
                float dx = zv.x - c.x, dy = zv.y - c.y;
                float dz = zv.z - c.z, dw = zv.w - c.w;
                lossAcc += dx * dx + dy * dy + dz * dz + dw * dw;
            }
        }
        __syncthreads();
    }

    // ---- deterministic loss reduction: wave -> WG -> d_ws partial ----
    #pragma unroll
    for (int m = 1; m < 32; m <<= 1) lossAcc += __shfl_xor(lossAcc, m, 32);
    if (lane == 0) lossW[wid] = lossAcc;
    __syncthreads();
    if (tid == 0) {
        float s = 0.f;
        #pragma unroll
        for (int w = 0; w < WAVES; ++w) s += lossW[w];
        wgPartial[blockIdx.x] = s;
    }
}

__global__ void vq_loss_finish(const float* __restrict__ wgPartial,
                               float* __restrict__ out)
{
    if (threadIdx.x == 0 && blockIdx.x == 0) {
        float s = 0.f;
        for (int i = 0; i < NUM_WGS; ++i) s += wgPartial[i];   // fixed order
        float m = s * (1.0f / (float)NTOT);
        out[NTOT]     = m;   // c_loss  (BETA = 1)
        out[NTOT + 1] = m;   // cb_loss (numerically identical)
    }
}

extern "C" void kernel_launch(void* const* d_in, const int* in_sizes, int n_in,
                              void* d_out, int out_size, void* d_ws, size_t ws_size,
                              hipStream_t stream)
{
    const float* z  = (const float*)d_in[0];   // [32,2048,64] fp32
    const float* cb = (const float*)d_in[1];   // [1024,64]    fp32
    float* out = (float*)d_out;                // 4194304 + 2 floats
    float* wgPartial = (float*)d_ws;           // 256 floats of scratch

    vq_main<<<NUM_WGS, THREADS, 0, stream>>>(z, cb, out, wgPartial);
    vq_loss_finish<<<1, 64, 0, stream>>>(wgPartial, out);
    (void)in_sizes; (void)n_in; (void)out_size; (void)ws_size;
}
